// SpatialConv_5884105195520
// MI455X (gfx1250) — compile-verified, matrix operand order checked by reading, not run on hardware
//
#include <hip/hip_runtime.h>
#include <hip/hip_bf16.h>

typedef _Float16 v16h __attribute__((ext_vector_type(16)));
typedef _Float16 v8h  __attribute__((ext_vector_type(8)));
typedef float    v8f  __attribute__((ext_vector_type(8)));

#define C_CH 64
#define Wd 256
#define Hd 256
#define LDS_STRIDE 72   /* halves per column (64 + 8 pad): 36*n mod 64 distinct -> conflict-free b128 */
#define NCOLS 258       /* 256 width + 2 zero halo columns */

#if __has_builtin(__builtin_amdgcn_tanhf)
__device__ __forceinline__ float fast_tanh(float x) { return __builtin_amdgcn_tanhf(x); }
#else
__device__ __forceinline__ float fast_tanh(float x) { return tanhf(x); }
#endif

// One workgroup per batch. 8 waves: wave -> (mt = co-tile 0..3, ntg = width half 0..1).
// Previous row kept in LDS (f16, transposed [col][ci]); scan runs serially over h inside
// the kernel with two workgroup barriers per step.
__global__ __launch_bounds__(256)
void spatial_conv_scan(const float* __restrict__ X,
                       const float* __restrict__ Wc,
                       const float* __restrict__ bc,
                       float* __restrict__ Y)
{
  __shared__ __align__(16) _Float16 Yt[NCOLS * LDS_STRIDE];

  const int tid  = threadIdx.x;
  const int lane = tid & 31;
  const int wave = tid >> 5;
  const int mt   = wave & 3;     // 16-wide block of output channels
  const int ntg  = wave >> 2;    // which 8 of the 16 width tiles
  const int b    = blockIdx.x;

  const int hiHalf = lane >> 4;  // 0: lanes 0-15, 1: lanes 16-31
  const int lmod   = lane & 15;

  // ---- zero the two halo columns ----
  for (int i = tid; i < 2 * LDS_STRIDE; i += 256) {
    int col = (i < LDS_STRIDE) ? 0 : (NCOLS - 1);
    Yt[col * LDS_STRIDE + (i % LDS_STRIDE)] = (_Float16)0.f;
  }

  // ---- row 0: Y[0] = X[0]; seed LDS ----
  const float* Xb = X + (size_t)b * C_CH * Hd * Wd;
  float*       Yb = Y + (size_t)b * C_CH * Hd * Wd;
  for (int i = tid; i < C_CH * Wd; i += 256) {
    int c = i >> 8, w = i & 255;
    float v = Xb[(size_t)c * Hd * Wd + w];
    Yb[(size_t)c * Hd * Wd + w] = v;
    Yt[(w + 1) * LDS_STRIDE + c] = (_Float16)v;
  }

  // ---- A fragments: weights, k-major (three K=64 GEMMs), per ISA 16-bit A 16x32 layout ----
  // lane holds row m = mt*16 + (lane&15); half j maps to K index:
  //   kk = j + (j>=8 ? 8 : 0) + (lane>=16 ? 8 : 0)
  v16h A[3][2];
  const int mrow = mt * 16 + lmod;
  #pragma unroll
  for (int k = 0; k < 3; ++k)
    #pragma unroll
    for (int ks = 0; ks < 2; ++ks)
      #pragma unroll
      for (int j = 0; j < 16; ++j) {
        int kk = j + ((j >= 8) ? 8 : 0) + (hiHalf ? 8 : 0);
        int ci = ks * 32 + kk;
        A[k][ks][j] = (_Float16)Wc[((size_t)mrow * C_CH + ci) * 3 + k];
      }

  // ---- bias per lane: this lane's 8 output rows are m0..m0+7 (C/D f32 layout) ----
  float bias[8];
  const int m0 = mt * 16 + hiHalf * 8;
  #pragma unroll
  for (int r = 0; r < 8; ++r) bias[r] = bc[m0 + r];

  __syncthreads();

  const int ci0 = hiHalf * 8;  // B-fragment: first 8 halves start at this ci (mirrors A striping)

  // Hoisted global pointers for this lane (advanced by Wd per step).
  // Per-tile offset i*16 and per-row offset r*Hd*Wd become immediate offsets.
  const float* xptr = Xb + (size_t)m0 * Hd * Wd + (size_t)Wd /*h=1*/ + ntg * 128 + lmod;
  float*       yptr = Yb + (size_t)m0 * Hd * Wd + (size_t)Wd /*h=1*/ + ntg * 128 + lmod;

  for (int h = 1; h < Hd; ++h) {
    v8h outv[8];
    #pragma unroll
    for (int i = 0; i < 8; ++i) {
      const int nt = ntg * 8 + i;
      const int w  = nt * 16 + lmod;   // width index this lane owns (N of the tile)
      v8f acc = {};
      #pragma unroll
      for (int k = 0; k < 3; ++k) {
        const int col = w + k;         // (w + k - 1) + 1 halo offset
        #pragma unroll
        for (int ks = 0; ks < 2; ++ks) {
          const _Float16* p = &Yt[col * LDS_STRIDE + ks * 32 + ci0];
          v8h lo = *(const v8h*)(p);        // ds_load_b128
          v8h hi = *(const v8h*)(p + 16);   // ds_load_b128
          v16h bf;
          #pragma unroll
          for (int j = 0; j < 8; ++j) { bf[j] = lo[j]; bf[j + 8] = hi[j]; }
          acc = __builtin_amdgcn_wmma_f32_16x16x32_f16(
              false, A[k][ks], false, bf, (short)0, acc, false, false);
        }
      }
      // epilogue: y = x + tanh(acc + bias); stream y out; stage f16 for next row
      const float* xp = xptr + i * 16;
      float*       yp = yptr + i * 16;
      __builtin_prefetch(xp + Wd, 0, 1);   // speculative prefetch of next X row (safe past end)
      #pragma unroll
      for (int r = 0; r < 8; ++r) {
        float x = xp[(size_t)r * Hd * Wd];
        float y = x + fast_tanh(acc[r] + bias[r]);
        yp[(size_t)r * Hd * Wd] = y;
        outv[i][r] = (_Float16)y;
      }
    }
    __syncthreads();   // all waves finished reading the old row
    #pragma unroll
    for (int i = 0; i < 8; ++i) {
      const int nt = ntg * 8 + i;
      const int w  = nt * 16 + lmod;
      *(v8h*)&Yt[(w + 1) * LDS_STRIDE + m0] = outv[i];   // ds_store_b128, conflict-free
    }
    __syncthreads();   // new row visible before next step
    xptr += Wd;
    yptr += Wd;
  }
}

extern "C" void kernel_launch(void* const* d_in, const int* in_sizes, int n_in,
                              void* d_out, int out_size, void* d_ws, size_t ws_size,
                              hipStream_t stream) {
  const float* X  = (const float*)d_in[0];   // [8,64,256,256] f32
  const float* Wc = (const float*)d_in[1];   // [64,64,3] f32
  const float* bc = (const float*)d_in[2];   // [64] f32
  float* Y = (float*)d_out;                  // [8,64,256,256] f32
  (void)in_sizes; (void)n_in; (void)out_size; (void)d_ws; (void)ws_size;
  spatial_conv_scan<<<dim3(8), dim3(256), 0, stream>>>(X, Wc, bc, Y);
}
